// LSTM_AE_20100446945939
// MI455X (gfx1250) — matrix operand order repaired and checked by reading, not compiled, not run on hardware
//
#include <hip/hip_runtime.h>
#include <hip/hip_bf16.h>

// ---------------------------------------------------------------------------
// LSTM_AE on MI455X (gfx1250).
// Encoder is dead code in the reference -> skipped entirely.
// Decoder = 1023 independent LSTM chains, 128 sequential steps each,
// D=H=256, shared weights. Per workgroup: tile of 16 chains, per step one
// (16,512)x(512,1024) bf16 WMMA GEMM, f32 accumulate, c in VGPRs, h in LDS.
// 16 waves/WG, one 16-wide hidden-unit group per wave (all 4 gate tiles of a
// group share the same C-matrix cell layout -> per-lane nonlinear update).
// ---------------------------------------------------------------------------

typedef __attribute__((ext_vector_type(16))) __bf16 v16bf;
typedef __attribute__((ext_vector_type(8)))  float  v8f;
typedef __attribute__((ext_vector_type(4)))  float  v4f;
typedef __attribute__((ext_vector_type(4)))  int    v4i;

union FragBF { v16bf bf; v4i i2[2]; };

#define SEQn    1024
#define BSTEPS  128      // scan length (batch axis of the torch module)
#define Dn      256
#define Hn      256
#define FOURH   1024
#define KTOT    512      // concat [x ; h]
#define RCHAINS 1023     // T-1 independent chains
#define M_TILE  16
#define THREADS 512      // 16 waves of 32
#define PACK_THREADS 256

__device__ __forceinline__ __bf16 f2bf(float f) {
  // round-to-nearest-even truncation f32 -> bf16 (bit-level, no libcalls)
  unsigned u = __builtin_bit_cast(unsigned, f);
  unsigned r = u + 0x7FFFu + ((u >> 16) & 1u);
  unsigned short s = (unsigned short)(r >> 16);
  return __builtin_bit_cast(__bf16, s);
}

// Branchless fast activations: 2 TRANS ops each (v_exp_f32 + v_rcp_f32),
// correct saturation at +/-inf, no exec-mask divergence like ocml tanhf.
__device__ __forceinline__ float fast_sigmoid(float x) {
  return __builtin_amdgcn_rcpf(1.0f + __expf(-x));
}
__device__ __forceinline__ float fast_tanh(float x) {
  return 1.0f - 2.0f * __builtin_amdgcn_rcpf(__expf(2.0f * x) + 1.0f);
}

// ---------------------------------------------------------------------------
// Pack kernel: Wc[n][k] bf16 (n=0..1023 gates, k=0..511 = [Wih_k | Whh_k]),
// combined bias (bih+bhh) f32, and zero out[0,:,:].
// ---------------------------------------------------------------------------
__global__ __launch_bounds__(PACK_THREADS) void pack_weights(
    const float* __restrict__ Wih, const float* __restrict__ Whh,
    const float* __restrict__ bih, const float* __restrict__ bhh,
    __bf16* __restrict__ Wc, float* __restrict__ biasC,
    float* __restrict__ out0) {
  int idx = blockIdx.x * blockDim.x + threadIdx.x;  // 0 .. 4H*KTOT-1
  int n = idx >> 9;          // gate row 0..1023
  int k = idx & 511;         // 0..511
  float v = (k < Dn) ? Wih[n * Dn + k] : Whh[n * Hn + (k - Dn)];
  Wc[idx] = f2bf(v);
  if (idx < FOURH)       biasC[idx] = bih[idx] + bhh[idx];
  if (idx < BSTEPS * Hn) out0[idx] = 0.0f;   // outputs[0] = zeros
}

// ---------------------------------------------------------------------------
// Persistent decoder kernel: one workgroup = 16 chains, loops 128 steps.
// Wave w owns hidden units j0 = w*16 .. +15 and the four gate tiles
// (i,f,g,o) at columns j0, 256+j0, 512+j0, 768+j0.
// ---------------------------------------------------------------------------
__global__ __launch_bounds__(THREADS) void lstm_dec_kernel(
    const float* __restrict__ src,      // (1024, 128, 256) f32
    const __bf16* __restrict__ Wc,      // (1024, 512) bf16
    const float* __restrict__ biasC,    // (1024,) f32
    float* __restrict__ out) {          // (1024, 128, 256) f32
  // act[m][0:256] = x_bf16 for this step, act[m][256:512] = h_bf16
  __shared__ __bf16 act[M_TILE * KTOT];   // 16 KB

  const int tid  = threadIdx.x;
  const int lane = tid & 31;
  const int wave = tid >> 5;     // 0..15
  const int half = lane >> 4;    // 0|1
  const int col  = lane & 15;    // C/D column, A row, B column
  const int r0   = blockIdx.x * M_TILE;
  const int j0   = wave * 16;    // this wave's hidden-unit group

  // Per-lane loop-invariants, hoisted by hand.
  // Weight addressing: uniform SGPR base (Wc) + 32-bit per-lane element
  // offsets -> saddr-form global_load_b128, no 64-bit VGPR pointer adds.
  float biasl[4];
  unsigned wo[4];
#pragma unroll
  for (int gate = 0; gate < 4; ++gate) {
    const int n = gate * Hn + j0 + col;
    biasl[gate] = biasC[n];
    wo[gate] = (unsigned)n * KTOT + half * 8;
  }
  // A-fragment base in LDS (16x32 bf16 layout: lanes 0-15 hold K 0-7/16-23,
  // lanes 16-31 hold K 8-15/24-31 of row M = lane&15).
  const unsigned ao = (unsigned)col * KTOT + half * 8;

  // Per-lane output bases and store guards, hoisted out of the b loop.
  // out element = ((r+1)*128 + b)*256 + j, r = r0 + m, m = v + half*8.
  size_t obase[8];
  bool   wr[8];
#pragma unroll
  for (int v = 0; v < 8; ++v) {
    int m = v + half * 8;
    int r = r0 + m;
    wr[v] = (r < RCHAINS);
    obase[v] = ((size_t)(r + 1) * BSTEPS) * Hn + (j0 + col);
  }

  // zero h region of act (h_0 = 0): 512 threads x 8 bf16 = 16x256
  {
    int m = tid >> 5, kb = lane * 8;
    unsigned* zp = (unsigned*)(act + m * KTOT + Dn + kb);
#pragma unroll
    for (int i = 0; i < 4; ++i) zp[i] = 0u;
  }

  float creg[8];
#pragma unroll
  for (int v = 0; v < 8; ++v) creg[v] = 0.0f;

  for (int b = 0; b < BSTEPS; ++b) {
    // ---- stage x tile: 16 chains x 256 f32 -> bf16 into act[:,0:256] ----
    {
      int m = tid >> 5, kb = lane * 8;
      const float* sp = src + ((size_t)(r0 + m) * BSTEPS + b) * Dn + kb;
      v4f x0 = *(const v4f*)(sp);
      v4f x1 = *(const v4f*)(sp + 4);
      unsigned* dp = (unsigned*)(act + m * KTOT + kb);
#pragma unroll
      for (int i = 0; i < 2; ++i) {
        unsigned short a0 = __builtin_bit_cast(unsigned short, f2bf(x0[2*i]));
        unsigned short a1 = __builtin_bit_cast(unsigned short, f2bf(x0[2*i+1]));
        dp[i] = (unsigned)a0 | ((unsigned)a1 << 16);
        unsigned short b0 = __builtin_bit_cast(unsigned short, f2bf(x1[2*i]));
        unsigned short b1 = __builtin_bit_cast(unsigned short, f2bf(x1[2*i+1]));
        dp[2 + i] = (unsigned)b0 | ((unsigned)b1 << 16);
      }
    }
    __syncthreads();

    // bias-initialized gate accumulators (i,f,g,o)
    v8f acc[4];
#pragma unroll
    for (int gate = 0; gate < 4; ++gate) {
#pragma unroll
      for (int v = 0; v < 8; ++v) acc[gate][v] = biasl[gate];
    }

    // K loop over concat [x ; h], 16 chunks of 32.
    // unroll_count(2) keeps the weight loads loop-variant (no LICM of the
    // whole 1MB weight set -> no scratch spilling), while still giving the
    // scheduler two chunks to software-pipeline loads under WMMAs.
#pragma clang loop unroll_count(2)
    for (int kc = 0; kc < KTOT; kc += 32) {
      FragBF A;
      A.i2[0] = *(const v4i*)(act + ao + kc);
      A.i2[1] = *(const v4i*)(act + ao + kc + 16);
#pragma unroll
      for (int gate = 0; gate < 4; ++gate) {
        FragBF B;  // 32x16 bf16 B layout: lane holds K 8h..8h+7 and 16+8h..16+8h+7 of col n
        B.i2[0] = *(const v4i*)(Wc + wo[gate] + kc);
        B.i2[1] = *(const v4i*)(Wc + wo[gate] + kc + 16);
        acc[gate] = __builtin_amdgcn_wmma_f32_16x16x32_bf16(
            false, A.bf, false, B.bf, (short)0, acc[gate], false, false);
      }
    }

    // per-lane LSTM cell update; c lives in VGPRs across all 128 steps
    float hreg[8];
#pragma unroll
    for (int v = 0; v < 8; ++v) {
      float iv = fast_sigmoid(acc[0][v]);
      float fv = fast_sigmoid(acc[1][v]);
      float gv = fast_tanh(acc[2][v]);
      float ov = fast_sigmoid(acc[3][v]);
      float cv = fv * creg[v] + iv * gv;
      creg[v] = cv;
      hreg[v] = ov * fast_tanh(cv);
    }
    __syncthreads();  // all reads of act for this step are done

    // write new h into act (bf16) and to global output (f32, streaming)
    {
      const int j = j0 + col;
      const size_t boff = (size_t)b * Hn;
#pragma unroll
      for (int v = 0; v < 8; ++v) {
        int m = v + half * 8;  // C-matrix row -> chain within tile
        act[m * KTOT + Dn + j] = f2bf(hreg[v]);
        if (wr[v])
          __builtin_nontemporal_store(hreg[v], &out[obase[v] + boff]);
      }
    }
    // next iteration: x staging touches only act[:,0:256]; the step-entry
    // barrier orders these h writes against the next step's fragment reads
  }
}

// ---------------------------------------------------------------------------
extern "C" void kernel_launch(void* const* d_in, const int* in_sizes, int n_in,
                              void* d_out, int out_size, void* d_ws, size_t ws_size,
                              hipStream_t stream) {
  (void)in_sizes; (void)n_in; (void)out_size; (void)ws_size;
  const float* src  = (const float*)d_in[0];
  // d_in[1..4] = encoder weights: dead code in the reference, never read.
  const float* dWih = (const float*)d_in[5];
  const float* dWhh = (const float*)d_in[6];
  const float* dbih = (const float*)d_in[7];
  const float* dbhh = (const float*)d_in[8];

  __bf16* Wc  = (__bf16*)d_ws;                                    // 1 MB
  float*  bC  = (float*)((char*)d_ws + (size_t)FOURH * KTOT * 2); // 4 KB
  float*  out = (float*)d_out;

  pack_weights<<<(FOURH * KTOT) / PACK_THREADS, PACK_THREADS, 0, stream>>>(
      dWih, dWhh, dbih, dbhh, Wc, bC, out);

  lstm_dec_kernel<<<(RCHAINS + M_TILE - 1) / M_TILE, THREADS, 0, stream>>>(
      src, Wc, bC, out);
}
